// MaskUnitAttention_59502476918804
// MI455X (gfx1250) — compile-verified, matrix operand order checked by reading, not run on hardware
//
#include <hip/hip_runtime.h>

// MaskUnitAttention fused for MI455X (gfx1250, wave32, WMMA bf16).
// ws layout: [w_qkv bf16: 1152*96][w_out bf16: 384*384][o_pre bf16: 65536*384]
// ws bytes needed: 221184 + 294912 + 50331648 = 50,847,744 (~48.5 MB)

typedef __attribute__((ext_vector_type(16))) __bf16        v16bf;
typedef __attribute__((ext_vector_type(8)))  float         v8f;
typedef __attribute__((ext_vector_type(8)))  unsigned int  v8u;

#define NH     8
#define DH     48
#define CIN    96
#define NTOK   1024
#define CHUNK  128
#define DMODEL 384

__device__ __forceinline__ unsigned short f2bf(float f) {
  unsigned int u = __builtin_bit_cast(unsigned int, f);
  return (unsigned short)((u + 0x7FFFu + ((u >> 16) & 1u)) >> 16);  // RNE
}

// K-offset inside a 32-wide K step for 16-bit A/B fragments (wave32):
// VGPR j in [0,4): K = 2j (+8 for lanes 16..31); j in [4,8): +16.
__device__ __forceinline__ int koffset(int j, int sub) {
  return ((j & 3) << 1) + (sub << 3) + ((j >> 2) << 4);
}

// Load one 16x32 (A) or 32x16 (B) bf16 fragment. rowptr already points at
// src + row*stride + kbase; consecutive-K elements must be contiguous.
__device__ __forceinline__ v16bf load_frag(const unsigned short* rowptr, int sub) {
  const unsigned int* p = (const unsigned int*)rowptr;
  v8u u;
#pragma unroll
  for (int j = 0; j < 8; ++j) u[j] = p[koffset(j, sub) >> 1];
  return __builtin_bit_cast(v16bf, u);
}

__device__ __forceinline__ v8f wmma_bf16(v16bf a, v16bf b, v8f c) {
  return __builtin_amdgcn_wmma_f32_16x16x32_bf16(false, a, false, b, (short)0, c,
                                                 false, false);
}

__global__ void cvt_bf16(const float* __restrict__ a, unsigned short* __restrict__ o, int n) {
  int i = blockIdx.x * 256 + threadIdx.x;
  if (i < n) o[i] = f2bf(a[i]);
}

// One block per image B (grid 256, 256 threads = 8 waves).
__global__ __launch_bounds__(256) void fused_attn(
    const float* __restrict__ x, const unsigned short* __restrict__ wq,
    unsigned short* __restrict__ opre) {
  __shared__ unsigned short xs[NTOK * CIN];      // [n][c]          192 KB
  __shared__ unsigned short wlds[3 * DH * CIN];  // [q|k|v][d][c]    27 KB
  __shared__ unsigned short es[DH * CHUNK];      // exp(k) [d][n]    12 KB
  __shared__ unsigned short vs[DH * CHUNK];      // v      [e][n]    12 KB
  __shared__ float          qs[DH * CHUNK];      // q      [d][n]    24 KB
  __shared__ float          ssum[DH];
  __shared__ unsigned short ctxb[DH * 64];       // ctx^T  [e][d pad 64]
  __shared__ unsigned short qsoft[32 * 64];      // [tok][d pad 64]

  const int tid  = threadIdx.x;
  const int wave = tid >> 5;
  const int lane = tid & 31;
  const int sub  = lane >> 4;   // which 16-lane half
  const int lm   = lane & 15;
  const int bimg = blockIdx.x;

  const float* xb = x + (size_t)bimg * CIN * NTOK;
  for (int i = tid; i < CIN * NTOK; i += 256) {       // coalesced over n
    int c = i >> 10, n = i & 1023;
    xs[n * CIN + c] = f2bf(xb[i]);
  }
  for (int i = tid; i < DH * 64; i += 256) if ((i & 63) >= DH) ctxb[i] = 0;
  for (int i = tid; i < 32 * 64; i += 256) if ((i & 63) >= DH) qsoft[i] = 0;

  // context tile ownership: wave w -> tile w of 3x3; wave 7 also tile 8
  const int di0 = wave / 3, ei0 = wave % 3;

  for (int h = 0; h < NH; ++h) {
    // stage this head's q/k/v weight slices (3 x 48 x 96 bf16) into LDS
    {
      const unsigned int* gw = (const unsigned int*)wq;
      unsigned int* lw = (unsigned int*)wlds;
      const int hw = h * (DH * CIN / 2);                // dword offset of head
      for (int i = tid; i < 3 * DH * CIN / 2; i += 256) {
        int mat = i / (DH * CIN / 2);
        int rem = i - mat * (DH * CIN / 2);
        lw[i] = gw[mat * (DMODEL * CIN / 2) + hw + rem];
      }
    }
    if (tid < DH) ssum[tid] = 0.0f;
    v8f cacc0 = {};
    v8f cacc1 = {};
    __syncthreads();

    const unsigned short* wqh = wlds;                   // [d][c]
    const unsigned short* wk  = wlds + DH * CIN;
    const unsigned short* wv  = wlds + 2 * DH * CIN;

    // ---- pass 1: k (exp) and v, accumulate context = E · V^T over n ----
    for (int ch = 0; ch < 8; ++ch) {
      const int n0   = ch * CHUNK;
      const int nloc = (wave << 4) + lm;              // wave owns one 16-col tile
      v16bf bx[3];
#pragma unroll
      for (int ks = 0; ks < 3; ++ks)                  // shared by all strips, k & v
        bx[ks] = load_frag(xs + (n0 + nloc) * CIN + (ks << 5), sub);
#pragma unroll
      for (int strip = 0; strip < 3; ++strip) {
        const int m0 = strip << 4;
        v8f ka = {}, va = {};
#pragma unroll
        for (int ks = 0; ks < 3; ++ks) {              // K = 96 = 3 x 32
          const int kb = ks << 5;
          v16bf ak = load_frag(wk + (m0 + lm) * CIN + kb, sub);
          v16bf av = load_frag(wv + (m0 + lm) * CIN + kb, sub);
          ka = wmma_bf16(ak, bx[ks], ka);
          va = wmma_bf16(av, bx[ks], va);
        }
#pragma unroll
        for (int r = 0; r < 8; ++r) {                 // D: m = m0+r+8*sub, n = nloc
          const int m = m0 + r + (sub << 3);
          float ev = __expf(ka[r]);                   // shift-free softmax numerator
          atomicAdd(&ssum[m], ev);
          es[m * CHUNK + nloc] = f2bf(ev);
          vs[m * CHUNK + nloc] = f2bf(va[r]);
        }
      }
      __syncthreads();
#pragma unroll
      for (int ks = 0; ks < 4; ++ks) {                // context: K = 128 = 4 x 32
        const int kb = ks << 5;
        v16bf a = load_frag(es + (di0 * 16 + lm) * CHUNK + kb, sub);
        v16bf b = load_frag(vs + (ei0 * 16 + lm) * CHUNK + kb, sub);
        cacc0 = wmma_bf16(a, b, cacc0);
      }
      if (wave == 7) {                                // extra tile (2,2)
#pragma unroll
        for (int ks = 0; ks < 4; ++ks) {
          const int kb = ks << 5;
          v16bf a = load_frag(es + (32 + lm) * CHUNK + kb, sub);
          v16bf b = load_frag(vs + (32 + lm) * CHUNK + kb, sub);
          cacc1 = wmma_bf16(a, b, cacc1);
        }
      }
      __syncthreads();
    }
    // finalize: ctxb[e][d] = ctx[d][e] / ssum[d]  (stored transposed for o-GEMM B)
#pragma unroll
    for (int r = 0; r < 8; ++r) {
      const int d = di0 * 16 + r + (sub << 3);
      const int e = ei0 * 16 + lm;
      ctxb[e * 64 + d] = f2bf(cacc0[r] / ssum[d]);
    }
    if (wave == 7) {
#pragma unroll
      for (int r = 0; r < 8; ++r) {
        const int d = 32 + r + (sub << 3);
        ctxb[(32 + lm) * 64 + d] = f2bf(cacc1[r] / ssum[d]);
      }
    }
    __syncthreads();

    // ---- pass 2: q, 2x2 max-pool, softmax over d, o = q_soft · ctx ----
    for (int ch = 0; ch < 8; ++ch) {                  // 128 cols = 4 spatial rows
      const int n0   = ch * CHUNK;
      const int nloc = (wave << 4) + lm;
      v16bf bx[3];
#pragma unroll
      for (int ks = 0; ks < 3; ++ks)
        bx[ks] = load_frag(xs + (n0 + nloc) * CIN + (ks << 5), sub);
#pragma unroll
      for (int strip = 0; strip < 3; ++strip) {
        const int m0 = strip << 4;
        v8f qa = {};
#pragma unroll
        for (int ks = 0; ks < 3; ++ks) {
          v16bf aq = load_frag(wqh + (m0 + lm) * CIN + (ks << 5), sub);
          qa = wmma_bf16(aq, bx[ks], qa);
        }
#pragma unroll
        for (int r = 0; r < 8; ++r)
          qs[(m0 + r + (sub << 3)) * CHUNK + nloc] = qa[r];
      }
      __syncthreads();
      if (tid < 32) {                                 // 32 pooled tokens this chunk
        const int pr = tid >> 4, px = tid & 15;
        const int c00 = pr * 64 + px * 2;
        float pv[DH];
        float mx = -3.0e38f;
#pragma unroll
        for (int d = 0; d < DH; ++d) {
          const float* qr = qs + d * CHUNK + c00;
          float v = fmaxf(fmaxf(qr[0], qr[1]), fmaxf(qr[32], qr[33]));
          pv[d] = v;
          mx = fmaxf(mx, v);
        }
        float s = 0.0f;
#pragma unroll
        for (int d = 0; d < DH; ++d) { pv[d] = __expf(pv[d] - mx); s += pv[d]; }
        const float inv = 0.14433756729740643f / s;   // d_head^-0.5 / sum
#pragma unroll
        for (int d = 0; d < DH; ++d) qsoft[tid * 64 + d] = f2bf(pv[d] * inv);
      }
      __syncthreads();
      if (wave < 6) {                                 // 2x3 output tiles (32x48)
        const int mi = wave / 3, ei = wave % 3;
        v8f oa = {};
#pragma unroll
        for (int ks = 0; ks < 2; ++ks) {              // K padded 48 -> 64
          const int kb = ks << 5;
          v16bf a = load_frag(qsoft + (mi * 16 + lm) * 64 + kb, sub);
          v16bf b = load_frag(ctxb + (ei * 16 + lm) * 64 + kb, sub);
          oa = wmma_bf16(a, b, oa);
        }
        const size_t rowbase = (size_t)bimg * 256 + ch * 32;
        const int col = h * DH + ei * 16 + lm;
#pragma unroll
        for (int r = 0; r < 8; ++r) {
          const size_t row = rowbase + mi * 16 + r + (sub << 3);
          opre[row * DMODEL + col] = f2bf(oa[r]);
        }
      }
      __syncthreads();
    }
  }
}

// out = o_pre(65536x384) · w_out^T + b_out, scattered to (B,384,16,16) layout.
// Block computes 64 rows x 384 cols; grid 1024.
__global__ __launch_bounds__(256) void proj_out(
    const unsigned short* __restrict__ opre, const unsigned short* __restrict__ wo,
    const float* __restrict__ bout, float* __restrict__ out) {
  const int tid = threadIdx.x;
  const int wave = tid >> 5, lane = tid & 31, sub = lane >> 4, lm = lane & 15;
  const size_t row0 = (size_t)blockIdx.x * 64;
  const int mi = wave & 3;
  const unsigned short* arow = opre + (row0 + mi * 16 + lm) * DMODEL;
  for (int nj = (wave >> 2); nj < 24; nj += 2) {
    v8f acc = {};
    const unsigned short* brow = wo + (nj * 16 + lm) * DMODEL;
#pragma unroll
    for (int ks = 0; ks < 12; ++ks) {                 // K = 384 = 12 x 32
      v16bf a = load_frag(arow + (ks << 5), sub);
      v16bf b = load_frag(brow + (ks << 5), sub);
      acc = wmma_bf16(a, b, acc);
    }
    const int o = nj * 16 + lm;
    const float bo = bout[o];
#pragma unroll
    for (int r = 0; r < 8; ++r) {
      const size_t rr = row0 + mi * 16 + r + (sub << 3);
      const size_t Bi = rr >> 8;                      // image index
      const int t = (int)(rr & 255);                  // pooled spatial 16*16
      out[(Bi * DMODEL + o) * 256 + t] = acc[r] + bo;
    }
  }
}

extern "C" void kernel_launch(void* const* d_in, const int* in_sizes, int n_in,
                              void* d_out, int out_size, void* d_ws, size_t ws_size,
                              hipStream_t stream) {
  const float* x    = (const float*)d_in[0];
  const float* wqkv = (const float*)d_in[1];
  const float* wout = (const float*)d_in[2];
  const float* bout = (const float*)d_in[3];

  unsigned short* wqkv_b = (unsigned short*)d_ws;
  unsigned short* wout_b = wqkv_b + 3 * DMODEL * CIN;           // 1152*96
  unsigned short* opre   = wout_b + DMODEL * DMODEL;            // 384*384
  float* out = (float*)d_out;

  cvt_bf16<<<(3 * DMODEL * CIN + 255) / 256, 256, 0, stream>>>(wqkv, wqkv_b,
                                                              3 * DMODEL * CIN);
  cvt_bf16<<<(DMODEL * DMODEL + 255) / 256, 256, 0, stream>>>(wout, wout_b,
                                                              DMODEL * DMODEL);
  fused_attn<<<256, 256, 0, stream>>>(x, wqkv_b, opre);
  proj_out<<<1024, 256, 0, stream>>>(opre, wout_b, bout, out);
}
